// Decoder_53291954209269
// MI455X (gfx1250) — matrix operand order repaired and checked by reading, not compiled, not run on hardware
//
#include <hip/hip_runtime.h>
#include <hip/hip_bf16.h>

// ---------------------------------------------------------------------------
// Pointer-network decoder for MI455X (gfx1250, wave32, WMMA).
//   B=128, S=256, H=512, Q=360 (padded to 384), N_HISTORY=3.
// Phases:
//   1) bf16 casts / padding / zero-init of history+mask       (prep kernels)
//   2) enc = inputs@Wc + bc   via v_wmma_f32_16x16x32_bf16    (17 GFLOP once)
//   3) per step: query=relu(prev@Wq) -> eq=query@w_q (WMMA)   (190 MFLOP/step)
//               score/sample kernel: v . tanh(enc+eq), Gumbel-max categorical,
//               log-softmax + entropy, mask & history update. enc (64MB) is
//               L2-resident (192MB L2), streamed with global_prefetch.
//   4) finalize: reduce log-probs/entropies, close the tour loop.
// GEMM blocks compute 128x64 tiles: each wave amortizes one A fragment over
// 4 WMMAs; A tile staged with GLOBAL_LOAD_ASYNC_TO_LDS_B128 (inline asm,
// ASYNCcnt-tracked) -- global->LDS with no VGPR round trip.
// ---------------------------------------------------------------------------

typedef __bf16 bf16_t;
typedef bf16_t v16bf __attribute__((ext_vector_type(16)));
typedef float  v8f   __attribute__((ext_vector_type(8)));

#define B_DIM   128
#define S_DIM   256
#define H_DIM   512
#define QP_DIM  384      // Q=360 padded to 384 (multiple of 64)
#define KH_DIM  1536     // N_HISTORY * H
#define TOUR_LD 257      // S+1

// ---- workspace layout (bytes) ---------------------------------------------
#define OFF_ENC    ((size_t)0)                          // f32 [32768 x 512]  = 64 MB
#define OFF_ABF    (OFF_ENC   + (size_t)67108864)       // bf16[32768 x 512]  = 32 MB
#define OFF_WCBF   (OFF_ABF   + (size_t)33554432)       // bf16[512 x 512]
#define OFF_WQBF   (OFF_WCBF  + (size_t)524288)         // bf16[1536 x 384]
#define OFF_WQ2BF  (OFF_WQBF  + (size_t)1179648)        // bf16[384 x 512]
#define OFF_PREV   (OFF_WQ2BF + (size_t)393216)         // bf16[128 x 1536]
#define OFF_QBF    (OFF_PREV  + (size_t)393216)         // bf16[128 x 384]
#define OFF_EQ     (OFF_QBF   + (size_t)98304)          // f32 [128 x 512]
#define OFF_MASK   (OFF_EQ    + (size_t)262144)         // f32 [128 x 256]
#define OFF_LPS    (OFF_MASK  + (size_t)131072)         // f32 [256 x 128]
#define OFF_ENTS   (OFF_LPS   + (size_t)131072)         // f32 [256 x 128]
// total ~104 MB

__device__ __forceinline__ unsigned short f2bf(float f) {
  unsigned u = __builtin_bit_cast(unsigned, f);
  unsigned r = u + 0x7FFFu + ((u >> 16) & 1u);   // round-to-nearest-even
  return (unsigned short)(r >> 16);
}

__device__ __forceinline__ unsigned pcg_hash(unsigned x) {
  x = x * 747796405u + 2891336453u;
  x = ((x >> ((x >> 28) + 4u)) ^ x) * 277803737u;
  return (x >> 22) ^ x;
}

// ---- CDNA5 async global->LDS copy (16B per lane), ASYNCcnt-tracked --------
// VDST = per-lane LDS byte address (low 32 bits of the generic LDS pointer,
// per the ISA aperture rule), VADDR = 64-bit global address, GV mode.
__device__ __forceinline__ void async_copy16_to_lds(const void* gsrc, void* ldst) {
  unsigned lds_addr = (unsigned)(unsigned long long)ldst;
  unsigned long long gaddr = (unsigned long long)gsrc;
  asm volatile("global_load_async_to_lds_b128 %0, %1, off"
               :: "v"(lds_addr), "v"(gaddr)
               : "memory");
}
__device__ __forceinline__ void wait_asynccnt0() {
  asm volatile("s_wait_asynccnt 0" ::: "memory");
}

// ---------------------------------------------------------------------------
// prep: inputs f32 -> bf16 (A matrix of the encoder GEMM)
// ---------------------------------------------------------------------------
__global__ __launch_bounds__(256)
void prep_inputs_kernel(const float* __restrict__ in, unsigned short* __restrict__ out) {
  size_t gid = (size_t)blockIdx.x * 256 + threadIdx.x;
  for (size_t i = gid; i < (size_t)B_DIM * S_DIM * H_DIM; i += (size_t)16384 * 256)
    out[i] = f2bf(in[i]);
}

// ---------------------------------------------------------------------------
// prep: weights -> bf16 (with zero padding), zero history + mask
// grid = 2304 x 256 covers 589824 = 1536*384 (largest range) exactly
// ---------------------------------------------------------------------------
__global__ __launch_bounds__(256)
void prep_misc_kernel(const float* __restrict__ Wc, const float* __restrict__ Wq,
                      const float* __restrict__ w_q,
                      unsigned short* __restrict__ WcBf, unsigned short* __restrict__ WqBf,
                      unsigned short* __restrict__ wq2Bf, unsigned short* __restrict__ prevBf,
                      float* __restrict__ mask) {
  int gid = blockIdx.x * 256 + threadIdx.x;
  if (gid < 512 * 512) WcBf[gid] = f2bf(Wc[gid]);
  {  // Wq [3,512,360] -> [1536,384] padded
    int r = gid / QP_DIM, c = gid - r * QP_DIM;
    WqBf[gid] = (c < 360) ? f2bf(Wq[r * 360 + c]) : (unsigned short)0;
  }
  if (gid < QP_DIM * H_DIM) {  // w_q [360,512] -> [384,512] padded
    int r = gid >> 9, c = gid & 511;
    wq2Bf[gid] = (r < 360) ? f2bf(w_q[r * 512 + c]) : (unsigned short)0;
  }
  if (gid < (B_DIM * KH_DIM) / 2) ((unsigned*)prevBf)[gid] = 0u;  // history = 0
  if (gid < B_DIM * S_DIM) mask[gid] = 0.0f;
}

// ---------------------------------------------------------------------------
// bf16 WMMA GEMM: C[M x N] = A[M x K] * B[K x N]  (M mult of 128, N mult of
// 64, K mult of 32). Block = 256 threads = 8 waves; block computes a 128x64
// tile. Wave w owns 16-row subtile w and all four 16-col subtiles -> one A
// fragment feeds 4 WMMAs. A tile staged async global->LDS.
// mode 0: f32 out + bias   mode 1: relu -> bf16 out   mode 2: f32 out
// ---------------------------------------------------------------------------
__global__ __launch_bounds__(256)
void wmma_gemm128_kernel(const unsigned short* __restrict__ A, int lda,
                         const unsigned short* __restrict__ B, int ldb,
                         void* __restrict__ Cout, int ldc, int K,
                         const float* __restrict__ bias, int mode) {
  __shared__ __align__(16) unsigned short ldsA[128 * 32];  // row-major 128x32 bf16
  __shared__ __align__(16) unsigned short ldsB[64 * 34];   // transposed [n][k], stride 34

  const int tid  = threadIdx.x;
  const int lane = tid & 31;
  const int wave = tid >> 5;
  const int m0   = blockIdx.y * 128;
  const int n0   = blockIdx.x * 64;
  const int ml   = lane & 15;
  const int half = lane >> 4;

  v8f acc0 = {}, acc1 = {}, acc2 = {}, acc3 = {};

#pragma unroll 2
  for (int kk = 0; kk < K; kk += 32) {
    // ---- stage A 128x32 bf16 (8 KB): 512 x 16B chunks, 2 per thread,
    //      async global->LDS (no VGPR round trip, ASYNCcnt-tracked) ---------
#pragma unroll
    for (int j = 0; j < 2; ++j) {
      int i = tid + j * 256;            // 0..511
      int row = i >> 2, c8 = (i & 3) * 8;
      async_copy16_to_lds(A + (size_t)(m0 + row) * lda + kk + c8,
                          &ldsA[row * 32 + c8]);
    }
    // ---- stage B 32x64 bf16, transposed into [n][k] ------------------------
#pragma unroll
    for (int j = 0; j < 4; ++j) {
      int i = tid + j * 256;            // 0..1023 dwords
      int krow = i >> 5, colu = i & 31;
      unsigned val = *(const unsigned*)(B + (size_t)(kk + krow) * ldb + n0 + colu * 2);
      ldsB[(colu * 2 + 0) * 34 + krow] = (unsigned short)(val & 0xFFFFu);
      ldsB[(colu * 2 + 1) * 34 + krow] = (unsigned short)(val >> 16);
    }
    wait_asynccnt0();
    __syncthreads();

    // ---- gather fragments per ISA layouts (16-bit A 16x32 / B 32x16) ------
    union { unsigned u[8]; v16bf v; } fa, fb0, fb1, fb2, fb3;
#pragma unroll
    for (int q = 0; q < 8; ++q) {
      int ka = ((q & 4) << 2) + half * 8 + ((q & 3) << 1);   // {0..7,16..23}+8*half
      fa.u[q] = *(const unsigned*)(&ldsA[(wave * 16 + ml) * 32 + ka]);
      int kb = ((q & 4) << 2) + half * 8 + ((q & 3) << 1);   // {0..7,16..23}+8*half
      fb0.u[q] = *(const unsigned*)(&ldsB[(0 * 16 + ml) * 34 + kb]);
      fb1.u[q] = *(const unsigned*)(&ldsB[(1 * 16 + ml) * 34 + kb]);
      fb2.u[q] = *(const unsigned*)(&ldsB[(2 * 16 + ml) * 34 + kb]);
      fb3.u[q] = *(const unsigned*)(&ldsB[(3 * 16 + ml) * 34 + kb]);
    }
    acc0 = __builtin_amdgcn_wmma_f32_16x16x32_bf16(false, fa.v, false, fb0.v, (short)0, acc0, false, false);
    acc1 = __builtin_amdgcn_wmma_f32_16x16x32_bf16(false, fa.v, false, fb1.v, (short)0, acc1, false, false);
    acc2 = __builtin_amdgcn_wmma_f32_16x16x32_bf16(false, fa.v, false, fb2.v, (short)0, acc2, false, false);
    acc3 = __builtin_amdgcn_wmma_f32_16x16x32_bf16(false, fa.v, false, fb3.v, (short)0, acc3, false, false);
    __syncthreads();
  }

  // ---- epilogue: C layout lanes0-15 -> M=q, lanes16-31 -> M=8+q, N=lane&15
#pragma unroll
  for (int ni = 0; ni < 4; ++ni) {
    v8f c = (ni == 0) ? acc0 : (ni == 1) ? acc1 : (ni == 2) ? acc2 : acc3;
    int col = n0 + ni * 16 + ml;
#pragma unroll
    for (int q = 0; q < 8; ++q) {
      int row = m0 + wave * 16 + half * 8 + q;
      float x = c[q];
      if (mode == 0) {
        ((float*)Cout)[(size_t)row * ldc + col] = x + bias[col];
      } else if (mode == 1) {
        ((unsigned short*)Cout)[(size_t)row * ldc + col] = f2bf(fmaxf(x, 0.0f));
      } else {
        ((float*)Cout)[(size_t)row * ldc + col] = x;
      }
    }
  }
}

// ---------------------------------------------------------------------------
// Per-step score + categorical sample + log-softmax + entropy + state update.
// One block per batch row b (grid 128), 256 threads = 8 waves.
// Phase 1: wave w scores rows s in [32w,32w+32); lanes coalesce over H.
// Phase 2: thread-per-s masking, Gumbel-max argmax, log-softmax, entropy.
// Phase 3: history shift + append bf16(action); mask update; tour write.
// ---------------------------------------------------------------------------
__global__ __launch_bounds__(256)
void score_sample_kernel(const float* __restrict__ enc, const float* __restrict__ eq,
                         const float* __restrict__ vvec, const float* __restrict__ inputs,
                         unsigned short* __restrict__ prevBf, float* __restrict__ mask,
                         float* __restrict__ tour, float* __restrict__ lps,
                         float* __restrict__ ents, int step) {
  __shared__ float s_eq[H_DIM];
  __shared__ float s_v[H_DIM];
  __shared__ float s_scores[S_DIM];
  __shared__ float red[S_DIM];
  __shared__ int   ridx[S_DIM];
  __shared__ float s_bcast;
  __shared__ int   s_choice;

  const int b    = blockIdx.x;
  const int tid  = threadIdx.x;
  const int lane = tid & 31;
  const int wave = tid >> 5;

  for (int h = tid; h < H_DIM; h += 256) { s_eq[h] = eq[b * H_DIM + h]; s_v[h] = vvec[h]; }
  __syncthreads();

  // ---- phase 1: raw scores  v . tanh(enc[b,s,:] + eq[b,:]) ----------------
  for (int j = 0; j < 32; ++j) {
    int s = wave * 32 + j;
    const float* e = enc + ((size_t)b * S_DIM + s) * H_DIM;
    if (j + 1 < 32) __builtin_prefetch(e + H_DIM + lane * 16, 0, 0);  // global_prefetch_b8
    float acc = 0.0f;
#pragma unroll
    for (int i = 0; i < 16; ++i) {
      int h = i * 32 + lane;                         // coalesced 128B per iter
      acc += s_v[h] * tanhf(e[h] + s_eq[h]);
    }
#pragma unroll
    for (int off = 16; off > 0; off >>= 1) acc += __shfl_down(acc, off, 32);
    if (lane == 0) s_scores[s] = acc;
  }
  __syncthreads();

  // ---- phase 2: mask, sample, log-softmax, entropy ------------------------
  const int s = tid;
  float sc = 10.0f * tanhf(s_scores[s]);                   // C_CLIP * tanh(score/TEMP)
  float masked = sc - 1.0e8f * mask[b * S_DIM + s];
  masked = fminf(fmaxf(masked, -1.0e8f), 1.0e8f);

  // Gumbel-max categorical sample (counter-based hash RNG)
  unsigned hsh = pcg_hash((unsigned)((step * B_DIM + b) * S_DIM + s) ^ 0x9E3779B9u);
  float u = fmaxf((float)(hsh >> 8) * (1.0f / 16777216.0f), 1.0e-7f);
  float g = -logf(-logf(u));

  red[s] = masked + g; ridx[s] = s;
  __syncthreads();
  for (int off = 128; off > 0; off >>= 1) {
    if (tid < off && red[tid + off] > red[tid]) { red[tid] = red[tid + off]; ridx[tid] = ridx[tid + off]; }
    __syncthreads();
  }
  if (tid == 0) s_choice = ridx[0];
  __syncthreads();
  const int idx = s_choice;

  red[s] = masked; __syncthreads();                         // block max
  for (int off = 128; off > 0; off >>= 1) {
    if (tid < off) red[tid] = fmaxf(red[tid], red[tid + off]);
    __syncthreads();
  }
  if (tid == 0) s_bcast = red[0];
  __syncthreads();
  const float maxv = s_bcast;

  red[s] = expf(masked - maxv); __syncthreads();            // block sum(exp)
  for (int off = 128; off > 0; off >>= 1) {
    if (tid < off) red[tid] += red[tid + off];
    __syncthreads();
  }
  if (tid == 0) s_bcast = logf(red[0]);
  __syncthreads();
  const float logZ = s_bcast;
  const float logp = masked - maxv - logZ;

  red[s] = -expf(logp) * logp; __syncthreads();             // entropy
  for (int off = 128; off > 0; off >>= 1) {
    if (tid < off) red[tid] += red[tid + off];
    __syncthreads();
  }

  if (tid == idx) { lps[step * B_DIM + b] = logp; mask[b * S_DIM + idx] = 1.0f; }
  if (tid == 0)   { ents[step * B_DIM + b] = red[0]; tour[b * TOUR_LD + step] = (float)idx; }
  __syncthreads();

  // ---- phase 3: history shift + append action (bf16, GEMM-ready) ----------
  unsigned tmp0 = *(const unsigned*)(&prevBf[b * KH_DIM + H_DIM + tid * 2]);
  unsigned tmp1 = *(const unsigned*)(&prevBf[b * KH_DIM + H_DIM + (tid + 256) * 2]);
  __syncthreads();
  *(unsigned*)(&prevBf[b * KH_DIM + tid * 2])         = tmp0;
  *(unsigned*)(&prevBf[b * KH_DIM + (tid + 256) * 2]) = tmp1;
  const float* act = inputs + ((size_t)b * S_DIM + idx) * H_DIM;
  for (int h = tid; h < H_DIM; h += 256)
    prevBf[b * KH_DIM + 2 * H_DIM + h] = f2bf(act[h]);
}

// ---------------------------------------------------------------------------
// finalize: sum log-probs / entropies, close tour loop
// ---------------------------------------------------------------------------
__global__ __launch_bounds__(256)
void finalize_kernel(float* __restrict__ out, const float* __restrict__ lps,
                     const float* __restrict__ ents) {
  __shared__ float r1[256], r2[256];
  int tid = threadIdx.x;
  float a = 0.0f, e = 0.0f;
  for (int i = tid; i < S_DIM * B_DIM; i += 256) { a += lps[i]; e += ents[i]; }
  r1[tid] = a; r2[tid] = e; __syncthreads();
  for (int off = 128; off > 0; off >>= 1) {
    if (tid < off) { r1[tid] += r1[tid + off]; r2[tid] += r2[tid + off]; }
    __syncthreads();
  }
  if (tid < B_DIM) out[tid * TOUR_LD + S_DIM] = out[tid * TOUR_LD];   // close loop
  if (tid == 0) { out[B_DIM * TOUR_LD] = r1[0]; out[B_DIM * TOUR_LD + 1] = r2[0]; }
}

// ---------------------------------------------------------------------------
extern "C" void kernel_launch(void* const* d_in, const int* in_sizes, int n_in,
                              void* d_out, int out_size, void* d_ws, size_t ws_size,
                              hipStream_t stream) {
  const float* inputs = (const float*)d_in[0];   // [128,256,512]
  const float* Wc     = (const float*)d_in[1];   // [512,512]
  const float* bc     = (const float*)d_in[2];   // [512]
  const float* Wq     = (const float*)d_in[3];   // [3,512,360]
  const float* w_q    = (const float*)d_in[4];   // [360,512]
  const float* vvec   = (const float*)d_in[5];   // [512]

  char* ws = (char*)d_ws;
  float*          enc    = (float*)(ws + OFF_ENC);
  unsigned short* Abf    = (unsigned short*)(ws + OFF_ABF);
  unsigned short* WcBf   = (unsigned short*)(ws + OFF_WCBF);
  unsigned short* WqBf   = (unsigned short*)(ws + OFF_WQBF);
  unsigned short* wq2Bf  = (unsigned short*)(ws + OFF_WQ2BF);
  unsigned short* prevBf = (unsigned short*)(ws + OFF_PREV);
  unsigned short* qBf    = (unsigned short*)(ws + OFF_QBF);
  float*          eqF    = (float*)(ws + OFF_EQ);
  float*          maskF  = (float*)(ws + OFF_MASK);
  float*          lpsF   = (float*)(ws + OFF_LPS);
  float*          entsF  = (float*)(ws + OFF_ENTS);
  float*          outF   = (float*)d_out;

  // 1) casts + padding + zero-init
  prep_inputs_kernel<<<16384, 256, 0, stream>>>(inputs, Abf);
  prep_misc_kernel<<<2304, 256, 0, stream>>>(Wc, Wq, w_q, WcBf, WqBf, wq2Bf, prevBf, maskF);

  // 2) encoder GEMM: enc[32768,512] = inputs@Wc + bc
  wmma_gemm128_kernel<<<dim3(H_DIM / 64, (B_DIM * S_DIM) / 128), 256, 0, stream>>>(
      Abf, H_DIM, WcBf, H_DIM, enc, H_DIM, H_DIM, bc, 0);

  // 3) sequential decode
  for (int step = 0; step < S_DIM; ++step) {
    // query = relu(prev @ Wq)  -> bf16 [128,384]
    wmma_gemm128_kernel<<<dim3(QP_DIM / 64, 1), 256, 0, stream>>>(
        prevBf, KH_DIM, WqBf, QP_DIM, qBf, QP_DIM, KH_DIM, nullptr, 1);
    // eq = query @ w_q  -> f32 [128,512]
    wmma_gemm128_kernel<<<dim3(H_DIM / 64, 1), 256, 0, stream>>>(
        qBf, QP_DIM, wq2Bf, H_DIM, eqF, H_DIM, QP_DIM, nullptr, 2);
    // attention scores + sample + state update
    score_sample_kernel<<<B_DIM, 256, 0, stream>>>(
        enc, eqF, vvec, inputs, prevBf, maskF, outF, lpsF, entsF, step);
  }

  // 4) reductions + tour closure
  finalize_kernel<<<1, 256, 0, stream>>>(outF, lpsF, entsF);
}
